// MultiHeadAttention_20650202759735
// MI455X (gfx1250) — compile-verified, matrix operand order checked by reading, not run on hardware
//
#include <hip/hip_runtime.h>
#include <hip/hip_bf16.h>

typedef __attribute__((ext_vector_type(16))) _Float16 v16h;
typedef __attribute__((ext_vector_type(8)))  _Float16 v8h;
typedef __attribute__((ext_vector_type(8)))  float    v8f;
typedef unsigned int v4u  __attribute__((ext_vector_type(4)));
typedef int          v8i_ __attribute__((ext_vector_type(8)));
typedef int          v4i_ __attribute__((ext_vector_type(4)));

static constexpr int Bc  = 4;
static constexpr int Tc  = 2048;
static constexpr int Dc  = 1024;
static constexpr int Hc  = 16;
static constexpr int HDc = 64;

// ---------------------------------------------------------------------------
// D = A*B + C  on v_wmma_f32_16x16x32_f16
// ---------------------------------------------------------------------------
__device__ __forceinline__ v8f wmma_f16(v16h a, v16h b, v8f c) {
  return __builtin_amdgcn_wmma_f32_16x16x32_f16(
      /*neg_a=*/false, a, /*neg_b=*/false, b,
      /*c_mod=*/(short)0, c, /*reuse_a=*/false, /*reuse_b=*/false);
}

// Load a 16x32 f16 A-fragment (or B-fragment: column-per-lane, same packing)
// from a row-major source with leading dimension `ld` (in halves).
// ISA layout: lanes 0-15 hold K=0..7 / 16..23, lanes 16-31 hold K=8..15 / 24..31.
__device__ __forceinline__ v16h load_frag(const _Float16* base, int ld) {
  const int lane = threadIdx.x & 31;
  const int r    = lane & 15;
  const int kb   = (lane >> 4) << 3;          // 0 or 8
  const _Float16* p = base + r * ld + kb;
  v8h lo = *(const v8h*)(p);                  // K = kb .. kb+7
  v8h hi = *(const v8h*)(p + 16);             // K = kb+16 .. kb+23
  v16h out;
#pragma unroll
  for (int i = 0; i < 8; ++i) { out[i] = lo[i]; out[i + 8] = hi[i]; }
  return out;
}

// ---------------------------------------------------------------------------
// Tensor Data Mover: 2D tile load Global -> LDS with LDS row padding.
// D# per CDNA5 ISA. Issued wave-uniform; TENSORcnt.
// pad_interval code c: pad fires every (2<<c) DWORDs; pad_amount code a: (a+1) DWORDs.
// ---------------------------------------------------------------------------
#if __has_builtin(__builtin_amdgcn_tensor_load_to_lds)
#define HAVE_TDM 1
__device__ __forceinline__ void tdm_load_2d(unsigned lds_addr, const void* gptr,
                                            unsigned tensor_d0, unsigned tensor_d1,
                                            unsigned d0_stride,
                                            unsigned tile_d0, unsigned tile_d1,
                                            unsigned pad_interval_c, unsigned pad_amount_c) {
  unsigned long long ga = (unsigned long long)gptr;
  v4u g0;
  g0[0] = 1u;                                        // count=1, user descriptor
  g0[1] = lds_addr;                                  // LDS byte address
  g0[2] = (unsigned)(ga & 0xFFFFFFFFu);              // global_addr[31:0]
  g0[3] = (unsigned)((ga >> 32) & 0x01FFFFFFu)       // global_addr[56:32]
        | (2u << 30);                                // type = 2 ("image")
  v8i_ g1;
  g1[0] = (int)((1u << 16)                           // data_size = 1 -> 2 bytes
              | (1u << 20)                           // pad_enable
              | (pad_interval_c << 22) | (pad_amount_c << 25));
  g1[1] = (int)((tensor_d0 & 0xFFFFu) << 16);        // tensor_dim0[15:0] @ bits63:48
  g1[2] = (int)((tensor_d0 >> 16) | ((tensor_d1 & 0xFFFFu) << 16));
  g1[3] = (int)((tensor_d1 >> 16) | (tile_d0 << 16));// tile_dim0 @ bits127:112
  g1[4] = (int)(tile_d1 & 0xFFFFu);                  // tile_dim1 (tile_dim2 = 0)
  g1[5] = (int)d0_stride;                            // tensor_dim0_stride[31:0]
  g1[6] = 0;
  g1[7] = 0;
  v4i_ z4 = {0, 0, 0, 0};
#if defined(__clang_major__) && (__clang_major__ >= 23)
  v8i_ z8 = {0, 0, 0, 0, 0, 0, 0, 0};
  __builtin_amdgcn_tensor_load_to_lds(g0, g1, z4, z4, z8, 0);
#else
  __builtin_amdgcn_tensor_load_to_lds(g0, g1, z4, z4, 0);
#endif
}
// Flat LDS address: hardware maps generic->LDS via addr[31:0] (ISA §10.2).
__device__ __forceinline__ unsigned lds_byte_addr(const void* p) {
  return (unsigned)(unsigned long long)p;
}
#else
#define HAVE_TDM 0
#endif

// ---------------------------------------------------------------------------
// GEMM: Out[M,N] = X[M,K] @ W[N,K]^T + bias[N]
// IN_F16 : 0 -> X is fp32 (converted to f16 in LDS), 1 -> X already f16
// OUT_MODE: 0 -> f16 out at [b][h][t][d]   (q, k)
//           1 -> f16 out at [b][h][d][t]   (v transposed for P@V fragments)
//           2 -> f32 out at [m][n]         (final output projection)
// ---------------------------------------------------------------------------
template <int IN_F16, int OUT_MODE>
__global__ __launch_bounds__(256)
void gemm_wmma_kernel(const void* __restrict__ Xv, const float* __restrict__ W,
                      const float* __restrict__ bias, void* __restrict__ Outv,
                      int M, int K, int N) {
  constexpr int BM = 128, BN = 64, BK = 32;
  constexpr int LDX = 40, LDW = 40;          // 80B row stride: 16B aligned, conflict-free
  __shared__ alignas(16) _Float16 sX[BM][LDX];
  __shared__ alignas(16) _Float16 sW[BN][LDW];

  const int tid  = threadIdx.x;
  const int lane = tid & 31;
  const int wave = tid >> 5;                 // 8 waves
  const int m0 = blockIdx.y * BM;
  const int n0 = blockIdx.x * BN;
  const int wm = (wave & 3) * 32;            // 4x2 wave grid, 32x32 per wave
  const int wn = (wave >> 2) * 32;

  v8f acc[2][2] = {};

  for (int k0 = 0; k0 < K; k0 += BK) {
    if constexpr (IN_F16) {
      const _Float16* X = (const _Float16*)Xv;
#pragma unroll
      for (int i = 0; i < 2; ++i) {          // 4096 halves = 512 v8h, 2/thread
        int idx = tid + i * 256;
        int row = idx >> 2;
        int c   = (idx & 3) * 8;
        *(v8h*)(&sX[row][c]) = *(const v8h*)(X + (size_t)(m0 + row) * K + k0 + c);
      }
    } else {
      const float* X = (const float*)Xv;
#pragma unroll
      for (int i = 0; i < 4; ++i) {          // 4096 f32 = 1024 float4, 4/thread
        int idx = tid + i * 256;
        int row = idx >> 3;
        int c   = (idx & 7) * 4;
        float4 v = *(const float4*)(X + (size_t)(m0 + row) * K + k0 + c);
        sX[row][c + 0] = (_Float16)v.x; sX[row][c + 1] = (_Float16)v.y;
        sX[row][c + 2] = (_Float16)v.z; sX[row][c + 3] = (_Float16)v.w;
      }
    }
#pragma unroll
    for (int i = 0; i < 2; ++i) {            // 2048 f32 = 512 float4, 2/thread
      int idx = tid + i * 256;
      int row = idx >> 3;
      int c   = (idx & 7) * 4;
      float4 v = *(const float4*)(W + (size_t)(n0 + row) * K + k0 + c);
      sW[row][c + 0] = (_Float16)v.x; sW[row][c + 1] = (_Float16)v.y;
      sW[row][c + 2] = (_Float16)v.z; sW[row][c + 3] = (_Float16)v.w;
    }
    __syncthreads();

    v16h a0 = load_frag(&sX[wm][0], LDX);
    v16h a1 = load_frag(&sX[wm + 16][0], LDX);
    v16h b0 = load_frag(&sW[wn][0], LDW);
    v16h b1 = load_frag(&sW[wn + 16][0], LDW);
    acc[0][0] = wmma_f16(a0, b0, acc[0][0]);
    acc[0][1] = wmma_f16(a0, b1, acc[0][1]);
    acc[1][0] = wmma_f16(a1, b0, acc[1][0]);
    acc[1][1] = wmma_f16(a1, b1, acc[1][1]);
    __syncthreads();
  }

  // Epilogue. C-layout: lane = N + 16*(M>=8), vgpr r -> M = r + 8*(lane>>4).
  const int rowoff = (lane >> 4) << 3;
#pragma unroll
  for (int i = 0; i < 2; ++i)
#pragma unroll
    for (int j = 0; j < 2; ++j)
#pragma unroll
      for (int r = 0; r < 8; ++r) {
        int Mi = m0 + wm + i * 16 + r + rowoff;
        int Ni = n0 + wn + j * 16 + (lane & 15);
        float val = acc[i][j][r] + bias[Ni];
        if constexpr (OUT_MODE == 2) {
          ((float*)Outv)[(size_t)Mi * N + Ni] = val;
        } else {
          int b = Mi / Tc, t = Mi % Tc;
          int h = Ni >> 6, d = Ni & 63;
          size_t idx;
          if constexpr (OUT_MODE == 0)
            idx = (((size_t)(b * Hc + h)) * Tc + t) * HDc + d;
          else
            idx = (((size_t)(b * Hc + h)) * HDc + d) * Tc + t;
          ((_Float16*)Outv)[idx] = (_Float16)val;
        }
      }
}

// ---------------------------------------------------------------------------
// Flash attention: one wave per 16 q-rows; WG = (b, h, 128 q-rows), 8 waves.
// Q/K f16 [B,H,T,64]; V f16 transposed [B,H,64,T]; online softmax in fp32.
// K/V tiles staged by the Tensor Data Mover into double-buffered LDS:
// wave0 issues tile i+1, s_wait_tensorcnt(2) guarantees tile i landed
// (TDM ops from one wave complete in order), barrier publishes it.
// ---------------------------------------------------------------------------
__global__ __launch_bounds__(256)
void attn_kernel(const _Float16* __restrict__ Q, const _Float16* __restrict__ Kf,
                 const _Float16* __restrict__ Vt, const float* __restrict__ abias,
                 const unsigned char* __restrict__ mask,
                 const float* __restrict__ hscale, _Float16* __restrict__ Ocat) {
  constexpr int LDK = 72, LDV = 40, LDP = 40;     // padded, 16B-aligned rows
  __shared__ alignas(16) _Float16 sK[2][32][LDK]; // K tile   [t][d], double-buffered
  __shared__ alignas(16) _Float16 sV[2][64][LDV]; // V^T tile [d][t], double-buffered
  __shared__ alignas(16) _Float16 sP[8][16][LDP]; // per-wave P (C->A relayout)

  const int tid = threadIdx.x, lane = tid & 31, wave = tid >> 5;
  const int nqb  = Tc / 128;
  const int bh   = blockIdx.x / nqb;
  const int qblk = blockIdx.x % nqb;
  const int b = bh / Hc, h = bh % Hc;
  const int q0 = qblk * 128 + wave * 16;
  const float scaling = 0.125f;                   // 1/sqrt(64)
  const int rowoff = (lane >> 4) << 3;
  const int nloc   = lane & 15;

  const _Float16* Kbh = Kf + (size_t)bh * Tc * HDc;   // [t][d], row stride 64
  const _Float16* Vbh = Vt + (size_t)bh * HDc * Tc;   // [d][t], row stride T

  // Q fragments for this wave's 16 rows (K = 0..31 and 32..63)
  const _Float16* qbase = Q + ((size_t)bh * Tc + q0) * HDc;
  v16h aq0 = load_frag(qbase, HDc);
  v16h aq1 = load_frag(qbase + 32, HDc);

  float mrun[8], lrun[8];
  v8f o[4] = {};
#pragma unroll
  for (int r = 0; r < 8; ++r) { mrun[r] = -__builtin_inff(); lrun[r] = 0.0f; }

#if HAVE_TDM
  if (wave == 0) {  // prologue: DMA tile 0 into buffer 0
    tdm_load_2d(lds_byte_addr(&sK[0][0][0]), Kbh,       64, 32, 64, 64, 32, 4, 3);
    tdm_load_2d(lds_byte_addr(&sV[0][0][0]), Vbh,       32, 64, Tc, 32, 64, 3, 3);
  }
#else
  const int krow = tid >> 3, kcol = (tid & 7) * 8;   // 32 x 64 halves
  const int vrow = tid >> 2, vcol = (tid & 3) * 8;   // 64 x 32 halves
#endif

  for (int kc = 0; kc < Tc; kc += 32) {
    const int cur = (kc >> 5) & 1;
    __syncthreads();   // everyone done reading buf[cur] (2 iters ago) and sP
#if HAVE_TDM
    if (wave == 0) {
      if (kc + 32 < Tc) {   // DMA next tile into the other buffer
        tdm_load_2d(lds_byte_addr(&sK[cur ^ 1][0][0]), Kbh + (size_t)(kc + 32) * HDc,
                    64, 32, 64, 64, 32, 4, 3);
        tdm_load_2d(lds_byte_addr(&sV[cur ^ 1][0][0]), Vbh + (kc + 32),
                    32, 64, Tc, 32, 64, 3, 3);
        __builtin_amdgcn_s_wait_tensorcnt(2);  // the 2 loads of buf[cur] are older
      } else {
        __builtin_amdgcn_s_wait_tensorcnt(0);  // drain
      }
    }
#else
    *(v8h*)(&sK[cur][krow][kcol]) =
        *(const v8h*)(Kbh + (size_t)(kc + krow) * HDc + kcol);
    *(v8h*)(&sV[cur][vrow][vcol]) =
        *(const v8h*)(Vbh + (size_t)vrow * Tc + kc + vcol);
#endif
    __syncthreads();   // publish buf[cur]

    // S = Q K^T : two 16x16 tiles, K-dim 64 = 2 WMMA steps each
    v8f s0 = {}, s1 = {};
    s0 = wmma_f16(aq0, load_frag(&sK[cur][0][0],   LDK), s0);
    s0 = wmma_f16(aq1, load_frag(&sK[cur][0][32],  LDK), s0);
    s1 = wmma_f16(aq0, load_frag(&sK[cur][16][0],  LDK), s1);
    s1 = wmma_f16(aq1, load_frag(&sK[cur][16][32], LDK), s1);

    // scale + bias + padding mask
#pragma unroll
    for (int r = 0; r < 8; ++r) {
      int q = q0 + r + rowoff;
      size_t broff = ((size_t)b * Tc + q) * Tc + kc;
      float x0 = s0[r] * scaling + abias[broff + nloc];
      float x1 = s1[r] * scaling + abias[broff + 16 + nloc];
      if (mask[b * Tc + kc + nloc])      x0 = -__builtin_inff();
      if (mask[b * Tc + kc + 16 + nloc]) x1 = -__builtin_inff();
      s0[r] = x0; s1[r] = x1;
    }

    // online softmax: row lives in one 16-lane half at one vgpr index
#pragma unroll
    for (int r = 0; r < 8; ++r) {
      float cm = fmaxf(s0[r], s1[r]);
#pragma unroll
      for (int off = 1; off < 16; off <<= 1) cm = fmaxf(cm, __shfl_xor(cm, off, 32));
      float mnew  = fmaxf(mrun[r], cm);
      float alpha = __expf(mrun[r] - mnew);
      mrun[r] = mnew;
      lrun[r] *= alpha;
      o[0][r] *= alpha; o[1][r] *= alpha; o[2][r] *= alpha; o[3][r] *= alpha;
      float p0 = __expf(s0[r] - mnew);
      float p1 = __expf(s1[r] - mnew);
      float ps = p0 + p1;
#pragma unroll
      for (int off = 1; off < 16; off <<= 1) ps += __shfl_xor(ps, off, 32);
      lrun[r] += ps;
      sP[wave][r + rowoff][nloc]      = (_Float16)p0;   // C-layout -> row-major
      sP[wave][r + rowoff][16 + nloc] = (_Float16)p1;
    }
    __syncthreads();   // order P stores before A-fragment reloads

    // O += P @ V  (P: 16x32 A-frag; V: 4 B-frags over d = 0..63)
    v16h pa = load_frag(&sP[wave][0][0], LDP);
#pragma unroll
    for (int n = 0; n < 4; ++n)
      o[n] = wmma_f16(pa, load_frag(&sV[cur][n * 16][0], LDV), o[n]);
  }

  // normalize, apply head scale, write f16 merged-head output [B,T,D]
  const float hs = hscale[h];
#pragma unroll
  for (int r = 0; r < 8; ++r) {
    int q = q0 + r + rowoff;
    float inv = hs / lrun[r];
    size_t obase = ((size_t)b * Tc + q) * Dc + h * HDc;
#pragma unroll
    for (int n = 0; n < 4; ++n)
      Ocat[obase + n * 16 + nloc] = (_Float16)(o[n][r] * inv);
  }
}

// ---------------------------------------------------------------------------
extern "C" void kernel_launch(void* const* d_in, const int* in_sizes, int n_in,
                              void* d_out, int out_size, void* d_ws, size_t ws_size,
                              hipStream_t stream) {
  (void)in_sizes; (void)n_in; (void)out_size; (void)ws_size;
  const float* query = (const float*)d_in[0];
  const float* key   = (const float*)d_in[1];
  const float* value = (const float*)d_in[2];
  const float* abias = (const float*)d_in[3];
  const float* ipw   = (const float*)d_in[4];
  const float* ipb   = (const float*)d_in[5];
  const float* outw  = (const float*)d_in[6];
  const float* outb  = (const float*)d_in[7];
  const float* hsc   = (const float*)d_in[8];
  const unsigned char* mask = (const unsigned char*)d_in[9];

  const int M = Bc * Tc;         // 8192 tokens
  const int K = Dc, N = Dc;

  char* ws = (char*)d_ws;
  const size_t one = (size_t)M * Dc * sizeof(_Float16);   // 16 MB each
  _Float16* qf = (_Float16*)(ws);
  _Float16* kf = (_Float16*)(ws + one);
  _Float16* vT = (_Float16*)(ws + 2 * one);
  _Float16* oc = (_Float16*)(ws + 3 * one);

  dim3 blk(256);
  dim3 g(N / 64, M / 128);
  gemm_wmma_kernel<0, 0><<<g, blk, 0, stream>>>((const void*)query, ipw,                       ipb,          (void*)qf, M, K, N);
  gemm_wmma_kernel<0, 0><<<g, blk, 0, stream>>>((const void*)key,   ipw + (size_t)Dc * Dc,     ipb + Dc,     (void*)kf, M, K, N);
  gemm_wmma_kernel<0, 1><<<g, blk, 0, stream>>>((const void*)value, ipw + 2 * (size_t)Dc * Dc, ipb + 2 * Dc, (void*)vT, M, K, N);

  dim3 ga(Bc * Hc * (Tc / 128));
  attn_kernel<<<ga, blk, 0, stream>>>(qf, kf, vT, abias, mask, hsc, oc);

  gemm_wmma_kernel<1, 2><<<g, blk, 0, stream>>>((const void*)oc, outw, outb, d_out, M, K, N);
}
